// LlamaAttention_34978213659023
// MI455X (gfx1250) — compile-verified, matrix operand order checked by reading, not run on hardware
//
#include <hip/hip_runtime.h>
#include <hip/hip_bf16.h>

// ---------------------------------------------------------------------------
// LLaMA attention prefill for MI455X (gfx1250, wave32, WMMA).
//   HIDDEN=2048, H=16, D=128, B=8, S=1024, T=8192
// bf16 WMMA everywhere; fragment-native layouts (2x16B contiguous loads per
// operand). GEMM uses 64x64 per-wave tiles -> 1:1 load:WMMA instruction ratio.
// ---------------------------------------------------------------------------

#define HIDDEN 2048
#define NHEADS 16
#define HDIM   128
#define BATCH  8
#define SEQ    1024
#define TTOK   (BATCH * SEQ)     // 8192
#define QKVN   (3 * HIDDEN)      // 6144

typedef __bf16 bf16;
typedef __attribute__((ext_vector_type(16))) __bf16 v16bf;
typedef __attribute__((ext_vector_type(8)))  __bf16 v8bf;
typedef __attribute__((ext_vector_type(8)))  float  v8f;

// Build a 16x16x32 bf16 WMMA operand from two contiguous 8-element (16B) runs.
// A-fragment: lane&15 = row M, (lane>>4)*8 = kb; elems 0..7 -> K=kb..kb+7,
// elems 8..15 -> K=16+kb..16+kb+7  (CDNA5 ISA 7.12.2).
static __device__ __forceinline__ v16bf load_frag_2x8(const bf16* p0, const bf16* p1) {
  v8bf lo = *(const v8bf*)p0;
  v8bf hi = *(const v8bf*)p1;
  v16bf r;
#pragma unroll
  for (int i = 0; i < 8; ++i) { r[i] = lo[i]; r[i + 8] = hi[i]; }
  return r;
}

static __device__ __forceinline__ v8f wmma_bf16(v16bf a, v16bf b, v8f c) {
  return __builtin_amdgcn_wmma_f32_16x16x32_bf16(false, a, false, b,
                                                 (short)0, c, false, false);
}

static __device__ __forceinline__ v8f v8f_zero() {
  v8f z;
#pragma unroll
  for (int i = 0; i < 8; ++i) z[i] = 0.0f;
  return z;
}

// ---------------------------------------------------------------------------
// Kernel 1: f32 -> bf16 conversion (vectorized 4/thread).
// ---------------------------------------------------------------------------
__global__ void cvt_f32_bf16(const float* __restrict__ s, bf16* __restrict__ d, int n) {
  int i = (blockIdx.x * blockDim.x + threadIdx.x) * 4;
  if (i + 3 < n) {
    float4 v = *(const float4*)(s + i);
    d[i + 0] = (bf16)v.x;
    d[i + 1] = (bf16)v.y;
    d[i + 2] = (bf16)v.z;
    d[i + 3] = (bf16)v.w;
  }
}

// ---------------------------------------------------------------------------
// Kernel 2: pack W[K,N] (f32 row-major) into WMMA B-fragment order:
//   frag f = ntile*(K/32)+ktile; 16 bf16 contiguous per lane:
//   n = ntile*16 + (lane&15), kb = (lane>>4)*8
//   elem i<8 -> W[ktile*32+kb+i][n], i>=8 -> W[ktile*32+16+kb+(i-8)][n]
// ---------------------------------------------------------------------------
__global__ void pack_w(const float* __restrict__ W, bf16* __restrict__ Bp,
                       int K, int N) {
  const int gid  = blockIdx.x * blockDim.x + threadIdx.x;
  const int f    = gid >> 5;
  const int lane = gid & 31;
  const int ktiles = K >> 5;
  if (f >= ktiles * (N >> 4)) return;
  const int nt = f / ktiles, kt = f % ktiles;
  const int n  = nt * 16 + (lane & 15);
  const int kb = (lane >> 4) * 8;
  bf16* out = Bp + ((size_t)f * 32 + lane) * 16;
#pragma unroll
  for (int i = 0; i < 8; ++i) {
    out[i]     = (bf16)W[(size_t)(kt * 32 + kb + i)      * N + n];
    out[i + 8] = (bf16)W[(size_t)(kt * 32 + 16 + kb + i) * N + n];
  }
}

// ---------------------------------------------------------------------------
// Kernel 3: GEMM C[M,ldc] = A[M,K](bf16) x Bpacked[K,N](bf16) with WMMA.
// 256 threads = 8 waves (2 M x 4 N); each wave owns a 64x64 block (4x4
// fragments). Per K-step: 8+8 b128 loads feed 16 WMMAs (1:1), maximizing
// matrix-pipe utilization; packed weights stay L2-resident (192 MB L2).
// ---------------------------------------------------------------------------
template <typename OutT>
__global__ __launch_bounds__(256) void gemm_wmma(const bf16* __restrict__ A,
                                                 const bf16* __restrict__ Bp,
                                                 OutT* __restrict__ C,
                                                 int K, int ldc) {
  const int lane  = threadIdx.x & 31;
  const int wid   = threadIdx.x >> 5;
  const int mbase = blockIdx.y * 128 + (wid >> 2) * 64;   // 2 waves in M
  const int nbase = blockIdx.x * 256 + (wid & 3) * 64;    // 4 waves in N
  const int row   = lane & 15;
  const int kb    = (lane >> 4) * 8;
  const int ktiles = K >> 5;

  v8f acc[4][4];
#pragma unroll
  for (int mt = 0; mt < 4; ++mt)
#pragma unroll
    for (int nt = 0; nt < 4; ++nt) acc[mt][nt] = v8f_zero();

  // Loop-invariant bases; per-fragment deltas become immediate offsets.
  const bf16* Abase = A + (size_t)(mbase + row) * K + kb;
  const bf16* Bbase = Bp + ((size_t)(nbase >> 4) * ktiles * 32 + lane) * 16;

  for (int kt = 0; kt < ktiles; ++kt) {
    const int k0 = kt * 32;
    v16bf a[4], b[4];
#pragma unroll
    for (int mt = 0; mt < 4; ++mt) {
      const bf16* p = Abase + (size_t)mt * 16 * K + k0;
      a[mt] = load_frag_2x8(p, p + 16);
    }
#pragma unroll
    for (int nt = 0; nt < 4; ++nt) {
      const bf16* p = Bbase + ((size_t)nt * ktiles + kt) * 32 * 16;
      b[nt] = *(const v16bf*)p;
    }
#pragma unroll
    for (int mt = 0; mt < 4; ++mt)
#pragma unroll
      for (int nt = 0; nt < 4; ++nt)
        acc[mt][nt] = wmma_bf16(a[mt], b[nt], acc[mt][nt]);
  }

  // C fragment: lane&15 = col N, VGPR r -> row r + 8*(lane>>4).
  const int half = lane >> 4;
  const int cn   = lane & 15;
#pragma unroll
  for (int mt = 0; mt < 4; ++mt)
#pragma unroll
    for (int nt = 0; nt < 4; ++nt)
#pragma unroll
      for (int r = 0; r < 8; ++r) {
        const size_t rr = (size_t)(mbase + mt * 16 + r + 8 * half);
        C[rr * ldc + (nbase + nt * 16 + cn)] = (OutT)acc[mt][nt][r];
      }
}

// ---------------------------------------------------------------------------
// Kernel 4: RoPE (in place on bf16 q,k; folds 1/sqrt(D) into q) + V
// transpose into vt[b,h,d,s] so P*V B-fragments load contiguous key runs.
// ---------------------------------------------------------------------------
__global__ __launch_bounds__(256) void rope_vt(bf16* __restrict__ qkv,
                                               bf16* __restrict__ vt,
                                               const int* __restrict__ context_lens) {
  const int t = blockIdx.x;
  const int b = t / SEQ, s = t % SEQ;
  const float pos = (float)(context_lens[b] + s);
  const float qscale = 0.08838834764831845f;  // 1/sqrt(128)
  for (int idx = threadIdx.x; idx < NHEADS * (HDIM / 2); idx += blockDim.x) {
    const int h = idx >> 6;
    const int d = idx & 63;
    const float inv = __powf(10000.0f, -(float)(2 * d) * (1.0f / (float)HDIM));
    float sn, cs;
    __sincosf(pos * inv, &sn, &cs);

    const size_t baseq = (size_t)t * QKVN + h * HDIM;
    float q1 = (float)qkv[baseq + d], q2 = (float)qkv[baseq + d + 64];
    qkv[baseq + d]      = (bf16)((q1 * cs - q2 * sn) * qscale);
    qkv[baseq + d + 64] = (bf16)((q1 * sn + q2 * cs) * qscale);

    const size_t basek = baseq + HIDDEN;
    float k1 = (float)qkv[basek + d], k2 = (float)qkv[basek + d + 64];
    qkv[basek + d]      = (bf16)(k1 * cs - k2 * sn);
    qkv[basek + d + 64] = (bf16)(k1 * sn + k2 * cs);

    const size_t basev = baseq + 2 * HIDDEN;
    const int bh = b * NHEADS + h;
    vt[((size_t)bh * HDIM + d)      * SEQ + s] = qkv[basev + d];
    vt[((size_t)bh * HDIM + d + 64) * SEQ + s] = qkv[basev + d + 64];
  }
}

// ---------------------------------------------------------------------------
// Kernel 5: causal flash attention, one wave per (b, h, 16-query tile).
// S = Q*K^T via WMMA; online softmax with half-wave shfl reductions; P staged
// through 1KB LDS to convert C-fragment -> A-fragment; P*V via WMMA on vt.
// Chunks past the causal diagonal are skipped entirely.
// ---------------------------------------------------------------------------
__global__ __launch_bounds__(32) void attn_kernel(const bf16* __restrict__ qkv,
                                                  const bf16* __restrict__ vt,
                                                  bf16* __restrict__ attn) {
  __shared__ __align__(16) bf16 Plds[16 * 32];

  const int lane  = threadIdx.x;
  const int qt    = blockIdx.x & 63;       // S/16 tiles
  const int bh    = blockIdx.x >> 6;
  const int b     = bh >> 4, h = bh & 15;
  const int qbase = qt * 16;               // local to sequence
  const int row   = lane & 15;
  const int half  = lane >> 4;
  const int kb    = half * 8;

  // Q A-fragments over D=128 (4 x K=32)
  v16bf qf[4];
  {
    const bf16* qp = qkv + (size_t)(b * SEQ + qbase + row) * QKVN + h * HDIM;
#pragma unroll
    for (int kk = 0; kk < 4; ++kk)
      qf[kk] = load_frag_2x8(qp + kk * 32 + kb, qp + kk * 32 + kb + 16);
  }

  float m[8], l[8];
  v8f o[8];
#pragma unroll
  for (int r = 0; r < 8; ++r) { m[r] = -3.0e38f; l[r] = 0.0f; }
#pragma unroll
  for (int dt = 0; dt < 8; ++dt) o[dt] = v8f_zero();

  const int nchunk = ((qbase + 15) >> 5) + 1;  // 32-key chunks up to diagonal

  for (int c = 0; c < nchunk; ++c) {
    const int kc = c * 32;

    // ---- S = Q * K^T : two 16-key tiles ----
    v8f st[2];
#pragma unroll
    for (int j = 0; j < 2; ++j) {
      const bf16* kp = qkv + (size_t)(b * SEQ + kc + j * 16 + row) * QKVN +
                       HIDDEN + h * HDIM;
      v8f sacc = v8f_zero();
#pragma unroll
      for (int kk = 0; kk < 4; ++kk) {
        v16bf kf = load_frag_2x8(kp + kk * 32 + kb, kp + kk * 32 + kb + 16);
        sacc = wmma_bf16(qf[kk], kf, sacc);
      }
      st[j] = sacc;
    }

    // ---- causal mask + online softmax ----
    const int cn = row;  // key col within tile
#pragma unroll
    for (int r = 0; r < 8; ++r) {
      const int qloc = qbase + r + 8 * half;
      if (kc + cn > qloc)      st[0][r] = -3.0e38f;
      if (kc + 16 + cn > qloc) st[1][r] = -3.0e38f;
    }

    float p0[8], p1[8], alpha[8];
#pragma unroll
    for (int r = 0; r < 8; ++r) {
      float mx = fmaxf(st[0][r], st[1][r]);
#pragma unroll
      for (int off = 8; off >= 1; off >>= 1)
        mx = fmaxf(mx, __shfl_xor(mx, off));     // stays in 16-lane half
      const float mn = fmaxf(m[r], mx);
      alpha[r] = __expf(m[r] - mn);
      m[r] = mn;
      p0[r] = __expf(st[0][r] - mn);
      p1[r] = __expf(st[1][r] - mn);
      float sum = p0[r] + p1[r];
#pragma unroll
      for (int off = 8; off >= 1; off >>= 1)
        sum += __shfl_xor(sum, off);
      l[r] = l[r] * alpha[r] + sum;
    }
#pragma unroll
    for (int dt = 0; dt < 8; ++dt)
#pragma unroll
      for (int r = 0; r < 8; ++r) o[dt][r] *= alpha[r];

    // ---- re-layout P (C-frag) -> A-frag via LDS ----
#pragma unroll
    for (int r = 0; r < 8; ++r) {
      Plds[(r + 8 * half) * 32 + cn]      = (bf16)p0[r];
      Plds[(r + 8 * half) * 32 + 16 + cn] = (bf16)p1[r];
    }
    __syncthreads();  // single wave: orders ds_store -> ds_load
    v16bf pf = load_frag_2x8(&Plds[row * 32 + kb], &Plds[row * 32 + 16 + kb]);

    // ---- O += P * V (vt is [bh, d, s]: contiguous key runs) ----
    const bf16* vbase = vt + (size_t)bh * HDIM * SEQ + kc;
#pragma unroll
    for (int dt = 0; dt < 8; ++dt) {
      const bf16* vp = vbase + (size_t)(dt * 16 + cn) * SEQ;
      v16bf vf = load_frag_2x8(vp + kb, vp + 16 + kb);
      o[dt] = wmma_bf16(pf, vf, o[dt]);
    }
    __syncthreads();
  }

  // ---- normalize and write attn (bf16, [T, H*D]) ----
  const int cn = row;
  bf16* op = attn + (size_t)(b * SEQ + qbase) * HIDDEN + h * HDIM;
#pragma unroll
  for (int r = 0; r < 8; ++r) {
    const float inv = 1.0f / l[r];
#pragma unroll
    for (int dt = 0; dt < 8; ++dt)
      op[(size_t)(r + 8 * half) * HIDDEN + dt * 16 + cn] = (bf16)(o[dt][r] * inv);
  }
}

// ---------------------------------------------------------------------------
// Launch
// ---------------------------------------------------------------------------
extern "C" void kernel_launch(void* const* d_in, const int* in_sizes, int n_in,
                              void* d_out, int out_size, void* d_ws, size_t ws_size,
                              hipStream_t stream) {
  const float* x            = (const float*)d_in[0];
  // d_in[1] = cu_seqlens (unused: equal-length packing)
  const int*   context_lens = (const int*)d_in[2];
  const float* wq           = (const float*)d_in[3];
  const float* wk           = (const float*)d_in[4];
  const float* wv           = (const float*)d_in[5];
  const float* wo           = (const float*)d_in[6];
  float*       out          = (float*)d_out;

  char* ws = (char*)d_ws;
  bf16* xb   = (bf16*)ws; ws += (size_t)TTOK * HIDDEN * 2;      // 32 MB
  bf16* pwq  = (bf16*)ws; ws += (size_t)HIDDEN * HIDDEN * 2;    // 8 MB
  bf16* pwk  = (bf16*)ws; ws += (size_t)HIDDEN * HIDDEN * 2;
  bf16* pwv  = (bf16*)ws; ws += (size_t)HIDDEN * HIDDEN * 2;
  bf16* pwo  = (bf16*)ws; ws += (size_t)HIDDEN * HIDDEN * 2;
  bf16* qkv  = (bf16*)ws; ws += (size_t)TTOK * QKVN * 2;        // 96 MB
  bf16* vt   = (bf16*)ws; ws += (size_t)TTOK * HIDDEN * 2;      // 32 MB
  bf16* attn = (bf16*)ws; ws += (size_t)TTOK * HIDDEN * 2;      // 32 MB

  // 1) x -> bf16
  {
    const int n = TTOK * HIDDEN;
    cvt_f32_bf16<<<dim3(n / (256 * 4)), dim3(256), 0, stream>>>(x, xb, n);
  }
  // 2) pack weights into B-fragment order (one-time)
  {
    const int frags = (HIDDEN / 32) * (HIDDEN / 16);             // 8192
    dim3 g(frags * 32 / 256), blk(256);
    pack_w<<<g, blk, 0, stream>>>(wq, pwq, HIDDEN, HIDDEN);
    pack_w<<<g, blk, 0, stream>>>(wk, pwk, HIDDEN, HIDDEN);
    pack_w<<<g, blk, 0, stream>>>(wv, pwv, HIDDEN, HIDDEN);
    pack_w<<<g, blk, 0, stream>>>(wo, pwo, HIDDEN, HIDDEN);
  }
  // 3) QKV projections -> qkv[T, 6144] bf16   (WG tile 128x256)
  {
    dim3 g(HIDDEN / 256, TTOK / 128), blk(256);
    gemm_wmma<bf16><<<g, blk, 0, stream>>>(xb, pwq, qkv + 0,          HIDDEN, QKVN);
    gemm_wmma<bf16><<<g, blk, 0, stream>>>(xb, pwk, qkv + HIDDEN,     HIDDEN, QKVN);
    gemm_wmma<bf16><<<g, blk, 0, stream>>>(xb, pwv, qkv + 2 * HIDDEN, HIDDEN, QKVN);
  }
  // 4) RoPE (in place on q,k) + V transpose
  rope_vt<<<dim3(TTOK), dim3(256), 0, stream>>>(qkv, vt, context_lens);
  // 5) causal flash attention
  attn_kernel<<<dim3(BATCH * NHEADS * (SEQ / 16)), dim3(32), 0, stream>>>(qkv, vt, attn);
  // 6) output projection -> f32 out
  {
    dim3 g(HIDDEN / 256, TTOK / 128), blk(256);
    gemm_wmma<float><<<g, blk, 0, stream>>>(attn, pwo, out, HIDDEN, HIDDEN);
  }
  (void)in_sizes; (void)n_in; (void)out_size; (void)ws_size;
}
